// GraphormerNeuralSDEModel_52003464020796
// MI455X (gfx1250) — compile-verified, matrix operand order checked by reading, not run on hardware
//
#include <hip/hip_runtime.h>
#include <hip/hip_bf16.h>

#define B_ 16
#define N_ 1024
#define E_ 16384
#define D_ 64
#define NBUCK 64

typedef __attribute__((ext_vector_type(16))) __bf16 bf16x16;
typedef __attribute__((ext_vector_type(8)))  __bf16 bf16x8;
typedef __attribute__((ext_vector_type(8)))  float  f32x8;

// K index mapping for 16-bit A/B fragments of v_wmma_*_16x16x32 (ISA 7.12.2):
// lane half 0: elems 0..7 -> K0..7, elems 8..15 -> K16..23
// lane half 1: elems 0..7 -> K8..15, elems 8..15 -> K24..31
__device__ __forceinline__ int kmap(int i, int half) {
  return (i < 8 ? i : i + 8) + 8 * half;
}

__device__ __forceinline__ f32x8 wmma_bf16(bf16x16 a, bf16x16 b, f32x8 c) {
  return __builtin_amdgcn_wmma_f32_16x16x32_bf16(false, a, false, b, (short)0, c,
                                                 false, false);
}

// ---------------------------------------------------------------- zero scratch
__global__ void k_zero(uint4* p, long n4) {
  long i = (long)blockIdx.x * blockDim.x + threadIdx.x;
  long stride = (long)gridDim.x * blockDim.x;
  uint4 z; z.x = 0u; z.y = 0u; z.z = 0u; z.w = 0u;
  for (; i < n4; i += stride) p[i] = z;
}

// --------------- fp32 weights -> bf16 in WMMA B-fragment-major order (Wq / 8)
// out[((nt*2+part)*32 + lane)*16 + i] = W[(part*32 + kmap(i,lane>>4))*64 + nt*16 + (lane&15)]
__global__ void k_prep_w(const float* wq, const float* wk, const float* wv,
                         const float* wo, __bf16* bq, __bf16* bk, __bf16* bv,
                         __bf16* bo) {
  int t = blockIdx.x * blockDim.x + threadIdx.x;   // 4 * 4096 threads
  int which = t >> 12, r = t & 4095;
  int i = r & 15, lane = (r >> 4) & 31, part = (r >> 9) & 1, nt = r >> 10;
  const float* s; __bf16* d; float sc = 1.0f;
  if (which == 0)      { s = wq; d = bq; sc = 0.125f; }  // fold 1/sqrt(64)
  else if (which == 1) { s = wk; d = bk; }
  else if (which == 2) { s = wv; d = bv; }
  else                 { s = wo; d = bo; }
  int src = (part * 32 + kmap(i, lane >> 4)) * D_ + nt * 16 + (lane & 15);
  d[r] = (__bf16)(s[src] * sc);
}

// ---------------------------------------------------------------- degree count
__global__ void k_deg(const int* esrc, const int* edst, int* ind, int* outd) {
  int t = blockIdx.x * blockDim.x + threadIdx.x;
  if (t >= B_ * E_) return;
  int b = t / E_;
  atomicAdd(&ind[b * N_ + edst[t]], 1);
  atomicAdd(&outd[b * N_ + esrc[t]], 1);
}

// --------------------------------------------------- dense edge bias scatter
__global__ void k_scatter(const int* esrc, const int* edst, const float* vals,
                          float* es) {
  int t = blockIdx.x * blockDim.x + threadIdx.x;
  if (t >= B_ * E_) return;
  int b = t / E_;
  es[(size_t)b * N_ * N_ + (size_t)esrc[t] * N_ + edst[t]] = vals[t];
}

// h = (emb[id] + in_emb[ib] + out_emb[ob]) * valid
// hf: plain fp32 [B*N][64] (residual).  hfrag: bf16 in A-fragment-major order:
// per 16-row block: [(blk*2+part)*32 + lane][i]  (inverse K-map below)
__global__ void k_h(const int* ids, const int* ind, const int* outd,
                    const float* emb, const float* ie, const float* oe,
                    float* hf, __bf16* hfrag) {
  int t = blockIdx.x * blockDim.x + threadIdx.x;   // B*N*64
  if (t >= B_ * N_ * D_) return;
  int d = t & 63, bn = t >> 6;
  int id = ids[bn];
  float v = (id > 0) ? 1.0f : 0.0f;
  int di = ind[bn], dox = outd[bn];
  int ib = (di  > 0) ? (31 - __clz(di))  : 0; if (ib > NBUCK - 1) ib = NBUCK - 1;
  int ob = (dox > 0) ? (31 - __clz(dox)) : 0; if (ob > NBUCK - 1) ob = NBUCK - 1;
  float val = (emb[(size_t)id * D_ + d] + ie[ib * D_ + d] + oe[ob * D_ + d]) * v;
  hf[t] = val;
  // inverse fragment map: d -> (part, K) ; K -> (half, i) ; row -> (blk, m)
  int blk = bn >> 4, m = bn & 15;
  int part = d >> 5, K = d & 31;
  int half = (K >> 3) & 1;
  int i = (K < 16) ? (K - 8 * half) : (K - 8 - 8 * half);
  int lane = m | (half << 4);
  hfrag[(((size_t)blk * 2 + part) * 32 + lane) * 16 + i] = (__bf16)val;
}

// q,k,v = h @ {Wq,Wk,Wv}.  A + B fragments are contiguous 32B vector loads.
// q,k stored in fragment-major order (via LDS transpose); v stored transposed
// vT[b][d][key] so attention V fragments are contiguous.
__global__ void __launch_bounds__(32)
k_qkv(const __bf16* hfrag, const __bf16* wqf, const __bf16* wkf,
      const __bf16* wvf, __bf16* qfrag, __bf16* kfrag, __bf16* vT) {
  __shared__ uint4 sT4[128];                       // 16x64 bf16 transpose tile
  __bf16* sT = (__bf16*)sT4;
  int lane = threadIdx.x;
  int blk = blockIdx.x;                            // 16-row block over B*N
  int m = lane & 15, half = lane >> 4;

  const bf16x16* hv = (const bf16x16*)hfrag;
  bf16x16 a_lo = hv[(blk * 2 + 0) * 32 + lane];
  bf16x16 a_hi = hv[(blk * 2 + 1) * 32 + lane];
  const bf16x16* Wf[3] = {(const bf16x16*)wqf, (const bf16x16*)wkf,
                          (const bf16x16*)wvf};
#pragma unroll
  for (int w = 0; w < 3; ++w) {
    f32x8 acc[4];
#pragma unroll
    for (int nt = 0; nt < 4; ++nt) {
      bf16x16 b_lo = Wf[w][(nt * 2 + 0) * 32 + lane];
      bf16x16 b_hi = Wf[w][(nt * 2 + 1) * 32 + lane];
      f32x8 a = {};
      a = wmma_bf16(a_hi, b_hi, a);
      a = wmma_bf16(a_lo, b_lo, a);
      acc[nt] = a;
    }
    __syncthreads();                               // sT free for reuse
#pragma unroll
    for (int nt = 0; nt < 4; ++nt)
#pragma unroll
      for (int r = 0; r < 8; ++r)                  // C/D layout: M = r + 8*half
        sT[(r + 8 * half) * D_ + nt * 16 + m] = (__bf16)acc[nt][r];
    __syncthreads();
    if (w < 2) {                                   // q / k -> fragment-major
      bf16x16 f_lo, f_hi;
#pragma unroll
      for (int i = 0; i < 16; ++i) {
        int kk = kmap(i, half);
        f_lo[i] = sT[m * D_ + kk];
        f_hi[i] = sT[m * D_ + 32 + kk];
      }
      bf16x16* dv = (bf16x16*)((w == 0) ? qfrag : kfrag);
      dv[(blk * 2 + 0) * 32 + lane] = f_lo;
      dv[(blk * 2 + 1) * 32 + lane] = f_hi;
    } else {                                       // v -> vT[b][d][key]
      int b = blk >> 6;
      int key0 = (blk & 63) * 16;
#pragma unroll
      for (int c = 0; c < 2; ++c) {
        int dd = lane + 32 * c;
        bf16x16 tcol;
#pragma unroll
        for (int j = 0; j < 16; ++j) tcol[j] = sT[j * D_ + dd];
        *(bf16x16*)(vT + ((size_t)b * D_ + dd) * N_ + key0) = tcol;
      }
    }
  }
}

// ------------- flash attention (+bias, +mask) fused with Wo GEMM + residual
__global__ void __launch_bounds__(32)
k_attn(const __bf16* qfrag, const __bf16* kfrag, const __bf16* vT,
       const float* es, const int* ids, const __bf16* wof, const float* hf,
       float* outf) {
  __shared__ uint4 sO4[128];    // 16x64 bf16: O staging for Wo GEMM
  __shared__ uint4 sP4[64];     // 16x32 bf16: P transpose
  __bf16* sO = (__bf16*)sO4; __bf16* sP = (__bf16*)sP4;

  int lane = threadIdx.x;
  int b = blockIdx.y;
  int q0 = blockIdx.x * 16;
  int blk = b * 64 + blockIdx.x;                   // global 16-row block id
  int m = lane & 15, half = lane >> 4;
  size_t base = ((size_t)b * N_ + q0) * D_;

  const bf16x16* qv = (const bf16x16*)qfrag;
  const bf16x16* kv = (const bf16x16*)kfrag;
  bf16x16 q_lo = qv[(blk * 2 + 0) * 32 + lane];
  bf16x16 q_hi = qv[(blk * 2 + 1) * 32 + lane];

  f32x8 O[4];
  float mrun[8], lrun[8];
#pragma unroll
  for (int nt = 0; nt < 4; ++nt) { f32x8 z = {}; O[nt] = z; }
#pragma unroll
  for (int r = 0; r < 8; ++r) { mrun[r] = -1e30f; lrun[r] = 0.0f; }

  for (int j0 = 0; j0 < N_; j0 += 32) {
    __syncthreads();                               // sP reuse fence

    // S = Q * K^T : K fragments are contiguous 32B/lane loads
    f32x8 S[2];
#pragma unroll
    for (int sub = 0; sub < 2; ++sub) {
      int t16 = b * 64 + (j0 >> 4) + sub;
      bf16x16 kb_lo = kv[(t16 * 2 + 0) * 32 + lane];
      bf16x16 kb_hi = kv[(t16 * 2 + 1) * 32 + lane];
      f32x8 acc = {};
      acc = wmma_bf16(q_hi, kb_hi, acc);
      acc = wmma_bf16(q_lo, kb_lo, acc);
      S[sub] = acc;
    }
    if (j0 + 32 < N_) {                            // gfx1250 global_prefetch_b8
      __builtin_prefetch(kfrag + ((size_t)(b * 64 + (j0 >> 4) + 2) * 64) * 16, 0, 3);
      __builtin_prefetch(vT + ((size_t)b * D_) * N_ + j0 + 32, 0, 3);
    }

    // bias + key mask (C/D layout: row = q0+r+8*half, col = j0+sub*16+m)
    float s0[8], s1[8];
#pragma unroll
    for (int sub = 0; sub < 2; ++sub) {
      int keyi = j0 + sub * 16 + m;
      bool kvalid = ids[b * N_ + keyi] > 0;
      float* sp = sub ? s1 : s0;
#pragma unroll
      for (int r = 0; r < 8; ++r) {
        int row = q0 + r + 8 * half;
        float sc = kvalid
            ? (S[sub][r] + es[((size_t)b * N_ + row) * N_ + keyi])
            : -1e9f;
        sp[r] = sc;
      }
    }

    // online softmax over the 32-key pair (16-lane row reductions)
#pragma unroll
    for (int r = 0; r < 8; ++r) {
      float t = fmaxf(s0[r], s1[r]);
#pragma unroll
      for (int mk = 1; mk < 16; mk <<= 1) t = fmaxf(t, __shfl_xor(t, mk, 32));
      float mn = fmaxf(mrun[r], t);
      float scale = __expf(mrun[r] - mn);
      mrun[r] = mn;
      lrun[r] *= scale;
      O[0][r] *= scale; O[1][r] *= scale; O[2][r] *= scale; O[3][r] *= scale;
      float p0 = __expf(s0[r] - mn), p1 = __expf(s1[r] - mn);
      float ps = p0 + p1;
#pragma unroll
      for (int mk = 1; mk < 16; mk <<= 1) ps += __shfl_xor(ps, mk, 32);
      lrun[r] += ps;
      int rm = r + 8 * half;                       // C-layout -> LDS P (16x32)
      sP[rm * 32 + m]      = (__bf16)p0;
      sP[rm * 32 + 16 + m] = (__bf16)p1;
    }
    __syncthreads();

    // O += P (16x32) * V (32x64); V fragments: two contiguous 16B loads/lane
    bf16x16 pA;
#pragma unroll
    for (int i = 0; i < 16; ++i) pA[i] = sP[m * 32 + kmap(i, half)];
#pragma unroll
    for (int nt = 0; nt < 4; ++nt) {
      const __bf16* vrow = vT + ((size_t)b * D_ + nt * 16 + m) * N_ + j0;
      bf16x8 lo8 = *(const bf16x8*)(vrow + 8 * half);
      bf16x8 hi8 = *(const bf16x8*)(vrow + 16 + 8 * half);
      bf16x16 vB;
#pragma unroll
      for (int i = 0; i < 8; ++i) { vB[i] = lo8[i]; vB[8 + i] = hi8[i]; }
      O[nt] = wmma_bf16(pA, vB, O[nt]);
    }
  }

  // normalize, stage O (bf16) into sO, then out = O @ Wo + h
  __syncthreads();
#pragma unroll
  for (int r = 0; r < 8; ++r) {
    float inv = 1.0f / lrun[r];
    int rm = r + 8 * half;
#pragma unroll
    for (int nt = 0; nt < 4; ++nt)
      sO[rm * D_ + nt * 16 + m] = (__bf16)(O[nt][r] * inv);
  }
  __syncthreads();

  bf16x16 o_lo, o_hi;
#pragma unroll
  for (int i = 0; i < 16; ++i) {
    int kk = kmap(i, half);
    o_lo[i] = sO[m * D_ + kk];
    o_hi[i] = sO[m * D_ + 32 + kk];
  }
  const bf16x16* wv_ = (const bf16x16*)wof;
#pragma unroll
  for (int nt = 0; nt < 4; ++nt) {
    bf16x16 w_lo = wv_[(nt * 2 + 0) * 32 + lane];
    bf16x16 w_hi = wv_[(nt * 2 + 1) * 32 + lane];
    f32x8 acc = {};
    acc = wmma_bf16(o_hi, w_hi, acc);
    acc = wmma_bf16(o_lo, w_lo, acc);
#pragma unroll
    for (int r = 0; r < 8; ++r) {
      int rm = r + 8 * half;
      size_t idx = base + (size_t)rm * D_ + nt * 16 + m;
      outf[idx] = acc[r] + hf[idx];                // residual
    }
  }
}

// ------------------- masked mean pools + predictor: relu(concat @ pw + pb)
__global__ void __launch_bounds__(64)
k_reduce(const float* outf, const int* ids, const float* ts, const float* pw,
         const float* pb, float* pred) {
  __shared__ float red[64];
  __shared__ float cnts[2];
  int b = blockIdx.x, d = threadIdx.x;

  float vs = 0.0f, tsm = 0.0f;
  for (int n = d; n < N_; n += 64) {
    int id = ids[b * N_ + n];
    float v = (id > 0) ? 1.0f : 0.0f;
    vs += v;
    float tl = ts[b * N_ + n] * v;
    tsm += (tl != 0.0f) ? 1.0f : 0.0f;
  }
  red[d] = vs; __syncthreads();
  for (int s = 32; s > 0; s >>= 1) { if (d < s) red[d] += red[d + s]; __syncthreads(); }
  if (d == 0) cnts[0] = fmaxf(red[0], 1.0f);
  __syncthreads();
  red[d] = tsm; __syncthreads();
  for (int s = 32; s > 0; s >>= 1) { if (d < s) red[d] += red[d + s]; __syncthreads(); }
  if (d == 0) cnts[1] = fmaxf(red[0], 1.0f);
  __syncthreads();

  float gs = 0.0f, ss = 0.0f;
  for (int n = 0; n < N_; ++n) {
    float o = outf[((size_t)b * N_ + n) * D_ + d];
    int id = ids[b * N_ + n];
    float v = (id > 0) ? 1.0f : 0.0f;
    float tl = ts[b * N_ + n] * v;
    gs += o * v;
    ss += o * ((tl != 0.0f) ? 1.0f : 0.0f);
  }
  float part = (gs / cnts[0]) * pw[d] + (ss / cnts[1]) * pw[64 + d];
  red[d] = part; __syncthreads();
  for (int s = 32; s > 0; s >>= 1) { if (d < s) red[d] += red[d + s]; __syncthreads(); }
  if (d == 0) pred[b] = fmaxf(red[0] + pb[0], 0.0f);
}

extern "C" void kernel_launch(void* const* d_in, const int* in_sizes, int n_in,
                              void* d_out, int out_size, void* d_ws,
                              size_t ws_size, hipStream_t stream) {
  const int*   node_ids = (const int*)d_in[0];
  const float* node_ts  = (const float*)d_in[1];
  const int*   edge_src = (const int*)d_in[2];
  const int*   edge_dst = (const int*)d_in[3];
  const float* evals    = (const float*)d_in[4];
  const float* emb      = (const float*)d_in[5];
  const float* ie       = (const float*)d_in[6];
  const float* oe       = (const float*)d_in[7];
  const float* wq       = (const float*)d_in[8];
  const float* wk       = (const float*)d_in[9];
  const float* wv       = (const float*)d_in[10];
  const float* wo       = (const float*)d_in[11];
  const float* pw       = (const float*)d_in[12];
  const float* pb       = (const float*)d_in[13];

  char* ws = (char*)d_ws;
  size_t off = 0;
  auto alloc = [&](size_t bytes) -> void* {
    void* p = ws + off;
    off = (off + bytes + 255) & ~(size_t)255;
    return p;
  };
  // zeroed region first (edge bias + degree counters, contiguous)
  float* es     = (float*)alloc((size_t)B_ * N_ * N_ * 4);   // 64 MB
  int*   ind    = (int*)  alloc((size_t)B_ * N_ * 4);
  int*   outd   = (int*)  alloc((size_t)B_ * N_ * 4);
  size_t zbytes = off;
  float*  hf    = (float*) alloc((size_t)B_ * N_ * D_ * 4);
  __bf16* hfr   = (__bf16*)alloc((size_t)B_ * N_ * D_ * 2);
  __bf16* qfr   = (__bf16*)alloc((size_t)B_ * N_ * D_ * 2);
  __bf16* kfr   = (__bf16*)alloc((size_t)B_ * N_ * D_ * 2);
  __bf16* vT    = (__bf16*)alloc((size_t)B_ * N_ * D_ * 2);
  float*  outf  = (float*) alloc((size_t)B_ * N_ * D_ * 4);
  __bf16* wqb   = (__bf16*)alloc(64 * 64 * 2);
  __bf16* wkb   = (__bf16*)alloc(64 * 64 * 2);
  __bf16* wvb   = (__bf16*)alloc(64 * 64 * 2);
  __bf16* wob   = (__bf16*)alloc(64 * 64 * 2);

  long n4 = (long)(zbytes / 16);
  k_zero<<<4096, 256, 0, stream>>>((uint4*)es, n4);
  k_prep_w<<<64, 256, 0, stream>>>(wq, wk, wv, wo, wqb, wkb, wvb, wob);
  k_deg<<<(B_ * E_ + 255) / 256, 256, 0, stream>>>(edge_src, edge_dst, ind, outd);
  k_scatter<<<(B_ * E_ + 255) / 256, 256, 0, stream>>>(edge_src, edge_dst, evals, es);
  k_h<<<(B_ * N_ * D_ + 255) / 256, 256, 0, stream>>>(node_ids, ind, outd, emb,
                                                      ie, oe, hf, hfr);
  k_qkv<<<(B_ * N_) / 16, 32, 0, stream>>>(hfr, wqb, wkb, wvb, qfr, kfr, vT);
  k_attn<<<dim3(N_ / 16, B_), 32, 0, stream>>>(qfr, kfr, vT, es, node_ids, wob,
                                               hf, outf);
  k_reduce<<<B_, 64, 0, stream>>>(outf, node_ids, node_ts, pw, pb, (float*)d_out);
}
